// SwinTransformerBlock_26955214749845
// MI455X (gfx1250) — matrix operand order repaired and checked by reading, not compiled, not run on hardware
//
#include <hip/hip_runtime.h>
#include <math.h>

typedef __bf16 bf16;
typedef __attribute__((ext_vector_type(16))) __bf16 v16bf;
typedef __attribute__((ext_vector_type(8)))  __bf16 v8bf;
typedef __attribute__((ext_vector_type(8)))  float  v8f;
typedef __attribute__((ext_vector_type(4)))  unsigned int v4u;
typedef __attribute__((ext_vector_type(8)))  int v8i;
typedef __attribute__((ext_vector_type(4)))  int v4i;

#define HEADS 6
#define HD    32
#define CDIM  192
#define NTOK  49
#define ROWS  128            // 2 windows * 64 padded rows
#define THREADS 256
#define LAMBDA_INIT_F 0.35550906759096926f   // 0.8 - 0.6*exp(-0.3)
#define QSCALE 0.17677669529663687f          // 32^-0.5

// packed-weight regions (bf16 element offsets in d_ws)
#define PK_QKV   0          // 36 chunks * (3072 hi + 3072 lo)
#define PK_PROJ  221184     //  6 chunks * (6144 hi + 6144 lo)
#define PK_FC1   294912     // 72 chunks * (2048 hi + 2048 lo)
#define PK_FC2   589824     // 24 chunks * (6144 hi + 6144 lo)
#define PK_TOTAL 884736     // bf16 elements
#define PK_BYTES ((size_t)PK_TOTAL * 2)

#if defined(__has_builtin)
#if __has_builtin(__builtin_amdgcn_tensor_load_to_lds) && __has_builtin(__builtin_amdgcn_s_wait_tensorcnt)
#define HAVE_TDM 1
#endif
#endif

// ---------------- small helpers ----------------

__device__ __forceinline__ v8f vzero8() {
  v8f z;
#pragma unroll
  for (int i = 0; i < 8; ++i) z[i] = 0.0f;
  return z;
}

__device__ __forceinline__ void split2(float x, bf16& h, bf16& l) {
  h = (bf16)x;
  l = (bf16)(x - (float)h);
}

// A-fragment (16-bit A 16x32): per-lane halves K = kb..kb+7 and kb+16..kb+23
__device__ __forceinline__ v16bf ldfragA(const bf16* p) {
  v8bf a = *(const v8bf*)(p);
  v8bf b = *(const v8bf*)(p + 16);
  v16bf r;
#pragma unroll
  for (int i = 0; i < 8; ++i) { r[i] = a[i]; r[i + 8] = b[i]; }
  return r;
}

// B-fragment (16-bit B 32x16): per-lane 16 contiguous K values
__device__ __forceinline__ v16bf ldfragB(const bf16* p) {
  return *(const v16bf*)p;
}

__device__ __forceinline__ float featf(float v) {
  return v >= 0.0f ? v + 1.0f : expf(v);
}

// window-pair row -> flat token index in (B, 56*56, C)
__device__ __forceinline__ int row2tok(int wpair, int r) {
  int w   = r >> 6;
  int n   = r & 63;
  int Wid = wpair * 2 + w;
  int b   = Wid >> 6;
  int rem = Wid & 63;
  int wy  = rem >> 3, wx = rem & 7;
  int ly  = n / 7,  lx = n % 7;
  return b * 3136 + (wy * 7 + ly) * 56 + (wx * 7 + lx);
}

// Fallback: stage a (nrows x 32) f32 weight chunk into LDS as bf16 hi/lo.
__device__ __forceinline__ void stageB(bf16* Bhi, bf16* Blo,
                                       const float* __restrict__ W, int ldw,
                                       int r0, int nrows, int noff, int k0, int tid) {
  int total = nrows * 32;
  for (int i = tid; i < total; i += THREADS) {
    int n = i >> 5, k = i & 31;
    size_t gidx = (size_t)(r0 + n) * ldw + (k0 + k);
    float v = W[gidx];
    if (k == 0 && (k0 + 32) < ldw)
      __builtin_prefetch(&W[gidx + 32], 0, 0);   // global_prefetch next K chunk
    bf16 h, l; split2(v, h, l);
    Bhi[(noff + n) * 32 + k] = h;
    Blo[(noff + n) * 32 + k] = l;
  }
}

#ifdef HAVE_TDM
// TDM: 1-D async copy of nelems bf16 from global to LDS (ISA 08_async_tensor D# layout).
__device__ __forceinline__ void issue_tdm(const bf16* gaddr, unsigned lds_byte_addr,
                                          unsigned nelems) {
  unsigned long long ga = (unsigned long long)(size_t)gaddr;
  v4u g0;
  g0[0] = 1u;                                               // count=1, user mode
  g0[1] = lds_byte_addr;                                    // lds_addr [63:32]
  g0[2] = (unsigned)ga;                                     // global_addr lo
  g0[3] = (unsigned)((ga >> 32) & 0x1FFFFFFu) | (2u << 30); // global_addr hi | type=2
  v8i g1;
  g1[0] = 0x10000;                                          // data_size=1 (2 bytes)
  g1[1] = (int)((nelems & 0xFFFFu) << 16);                  // tensor_dim0 [15:0]
  g1[2] = (int)((nelems >> 16) | (1u << 16));               // tensor_dim0 hi | tensor_dim1=1
  g1[3] = (int)((nelems & 0xFFFFu) << 16);                  // tile_dim0 = nelems
  g1[4] = 1;                                                // tile_dim1=1, tile_dim2=0
  g1[5] = (int)nelems;                                      // tensor_dim0_stride lo
  g1[6] = (int)((nelems & 0xFFFFu) << 16);                  // stride0 hi=0 | stride1 lo
  g1[7] = (int)(nelems >> 16);                              // stride1 hi
  v4i z4; z4[0] = 0; z4[1] = 0; z4[2] = 0; z4[3] = 0;
  v8i z8;
#pragma unroll
  for (int i = 0; i < 8; ++i) z8[i] = 0;
  __builtin_amdgcn_tensor_load_to_lds(g0, g1, z4, z4, z8, 0);
}

// Stage pre-packed hi/lo chunk: hi block at src, lo block at src+nelems.
__device__ __forceinline__ void tdm_stage(const bf16* src, unsigned offBhi, unsigned offBlo,
                                          unsigned nelems, int wave) {
  if (wave == 0) {
    issue_tdm(src,          offBhi, nelems);
    issue_tdm(src + nelems, offBlo, nelems);
    __builtin_amdgcn_s_wait_tensorcnt(0);
  }
}
#endif

// One 32-K WMMA step for NT 16x16 output tiles (split-bf16: 3 WMMAs / tile).
template <int NT>
__device__ __forceinline__ void gemm_step(const bf16* __restrict__ Ahi,
                                          const bf16* __restrict__ Alo,
                                          int lda, int m0, int k0,
                                          const bf16* __restrict__ Bhi,
                                          const bf16* __restrict__ Blo,
                                          v8f* acc, int lane) {
  const int kbA = (lane & 16) ? 16 : 0;
  const int kbB = (lane & 16) ? 16 : 0;
  const int ar  = m0 + (lane & 15);
  v16bf ah = ldfragB(Ahi + (size_t)ar * lda + k0 + kbA);
  v16bf al = ldfragB(Alo + (size_t)ar * lda + k0 + kbA);
#pragma unroll
  for (int t = 0; t < NT; ++t) {
    int bn = t * 16 + (lane & 15);
    v16bf bh = ldfragB(Bhi + bn * 32 + kbB);
    v16bf bl = ldfragB(Blo + bn * 32 + kbB);
    acc[t] = __builtin_amdgcn_wmma_f32_16x16x32_bf16(false, ah, false, bh, (short)0, acc[t], false, false);
    acc[t] = __builtin_amdgcn_wmma_f32_16x16x32_bf16(false, ah, false, bl, (short)0, acc[t], false, false);
    acc[t] = __builtin_amdgcn_wmma_f32_16x16x32_bf16(false, al, false, bh, (short)0, acc[t], false, false);
  }
}

// ---------------- prologue: repack weights into TDM-friendly bf16 hi/lo chunks ----------------

__global__ __launch_bounds__(THREADS)
void pack_weights(const float* __restrict__ qkvw, const float* __restrict__ projw,
                  const float* __restrict__ f1w,  const float* __restrict__ f2w,
                  bf16* __restrict__ pk) {
  int e = blockIdx.x * THREADS + threadIdx.x;
  float v; size_t dhi; int bs;
  if (e < 110592) {                       // qkv: [h][kc6] chunks of 96x32
    int blk = e / 3072, r = e % 3072;
    int h = blk / 6, kc6 = blk % 6;
    int n = r >> 5, k = r & 31;
    int p = n >> 5, d = n & 31;
    v   = qkvw[(size_t)(p * CDIM + h * HD + d) * CDIM + kc6 * 32 + k];
    dhi = PK_QKV + (size_t)blk * 6144 + r;  bs = 3072;
  } else if (e < 110592 + 36864) {        // proj: [kc6] chunks of 192x32
    int e2 = e - 110592;
    int blk = e2 / 6144, r = e2 % 6144;
    int n = r >> 5, k = r & 31;
    v   = projw[(size_t)n * CDIM + blk * 32 + k];
    dhi = PK_PROJ + (size_t)blk * 12288 + r;  bs = 6144;
  } else if (e < 110592 + 36864 + 147456) { // fc1: [kc2g][kc6] chunks of 64x32
    int e2 = e - (110592 + 36864);
    int blk = e2 / 2048, r = e2 % 2048;
    int kc2g = blk / 6, kc6 = blk % 6;
    int n = r >> 5, k = r & 31;
    v   = f1w[(size_t)(kc2g * 64 + n) * CDIM + kc6 * 32 + k];
    dhi = PK_FC1 + (size_t)blk * 4096 + r;  bs = 2048;
  } else if (e < PK_TOTAL / 2) {          // fc2: [kc2g][kkg] chunks of 192x32
    int e2 = e - (110592 + 36864 + 147456);
    int blk = e2 / 6144, r = e2 % 6144;
    int kc2g = blk / 2, kkg = blk % 2;
    int n = r >> 5, k = r & 31;
    v   = f2w[(size_t)n * (4 * CDIM) + kc2g * 64 + kkg * 32 + k];
    dhi = PK_FC2 + (size_t)blk * 12288 + r;  bs = 6144;
  } else {
    return;
  }
  bf16 h, l; split2(v, h, l);
  pk[dhi] = h;
  pk[dhi + bs] = l;
}

// ---------------- fused Swin differential-linear-attention block ----------------

__global__ __launch_bounds__(THREADS, 1)
void swin_diffattn_block(const float* __restrict__ x,
                         const float* __restrict__ n1w, const float* __restrict__ n1b,
                         const float* __restrict__ qkvw, const float* __restrict__ qkvb,
                         const float* __restrict__ projw, const float* __restrict__ projb,
                         const float* __restrict__ lk1, const float* __restrict__ lv1,
                         const float* __restrict__ lk2, const float* __restrict__ lv2,
                         const float* __restrict__ n2w, const float* __restrict__ n2b,
                         const float* __restrict__ f1w, const float* __restrict__ f1b,
                         const float* __restrict__ f2w, const float* __restrict__ f2b,
                         const bf16* __restrict__ wpk, int usepk,
                         float* __restrict__ out) {
  __shared__ bf16  sAhi[ROWS * CDIM];     // 48 KB   A operand (hi)
  __shared__ bf16  sAlo[ROWS * CDIM];     // 48 KB   A operand (lo)
  __shared__ float sF32[ROWS * CDIM];     // 96 KB   f32 scratch (x tile / attn-out / x1)
  __shared__ float sQKV[ROWS * 96];       // 48 KB   per-head q,k,v
  __shared__ bf16  sHhi[ROWS * 64];       // 16 KB   MLP hidden chunk (hi)
  __shared__ bf16  sHlo[ROWS * 64];       // 16 KB   MLP hidden chunk (lo)
  __shared__ bf16  sBhi[192 * 32];        // 12 KB   staged weights (hi)
  __shared__ bf16  sBlo[192 * 32];        // 12 KB   staged weights (lo)
  __shared__ float sATT[2 * HD * HD];     //  8 KB   per-window dxd attention
  __shared__ float sMU[ROWS];
  __shared__ float sRS[ROWS];
  __shared__ float sLAM[8];

  const int tid   = threadIdx.x;
  const int lane  = tid & 31;
  const int wave  = tid >> 5;
  const int wpair = blockIdx.x;           // windows 2*wpair, 2*wpair+1
  const int m0    = wave * 16;            // each wave owns one 16-row M tile
#ifdef HAVE_TDM
  const unsigned offBhi = (unsigned)(size_t)(void*)&sBhi[0];
  const unsigned offBlo = (unsigned)(size_t)(void*)&sBlo[0];
#endif

  // ---- lambda scales: (1 - (exp(lk1.lv1) - exp(lk2.lv2) + lambda_init)) per head
  if (tid < HEADS) {
    float s1 = 0.f, s2 = 0.f;
    for (int i = 0; i < HD; ++i) {
      s1 += lk1[tid * HD + i] * lv1[tid * HD + i];
      s2 += lk2[tid * HD + i] * lv2[tid * HD + i];
    }
    sLAM[tid] = 1.0f - (expf(s1) - expf(s2) + LAMBDA_INIT_F);
  }

  // ---- stage 1: gather window tokens into LDS (zeros in padded rows)
  for (int i = tid; i < ROWS * CDIM; i += THREADS) {
    int r = i / CDIM, c = i % CDIM;
    float v = 0.0f;
    if ((r & 63) < NTOK) v = x[(size_t)row2tok(wpair, r) * CDIM + c];
    sF32[i] = v;
  }
  __syncthreads();

  // ---- stage 2: LayerNorm1 -> split-bf16 A operand
  if (tid < ROWS) {
    float s = 0.f, ss = 0.f;
    for (int c = 0; c < CDIM; ++c) { float v = sF32[tid * CDIM + c]; s += v; ss += v * v; }
    float mu = s / CDIM;
    sMU[tid] = mu;
    sRS[tid] = 1.0f / sqrtf(ss / CDIM - mu * mu + 1e-5f);
  }
  __syncthreads();
  for (int i = tid; i < ROWS * CDIM; i += THREADS) {
    int r = i / CDIM, c = i % CDIM;
    float v = 0.0f;
    if ((r & 63) < NTOK) v = (sF32[i] - sMU[r]) * sRS[r] * n1w[c] + n1b[c];
    bf16 h, l; split2(v, h, l);
    sAhi[i] = h; sAlo[i] = l;
  }
  __syncthreads();

  // ---- stage 3: per-head QKV GEMM + differential linear attention
  for (int h = 0; h < HEADS; ++h) {
    v8f qacc[6];
#pragma unroll
    for (int t = 0; t < 6; ++t) qacc[t] = vzero8();

    for (int kc = 0; kc < CDIM; kc += 32) {
      __syncthreads();                                    // ldsB reuse fence
#ifdef HAVE_TDM
      if (usepk) {
        tdm_stage(wpk + PK_QKV + (size_t)(h * 6 + (kc >> 5)) * 6144, offBhi, offBlo, 3072, wave);
      } else
#endif
      {
        stageB(sBhi, sBlo, qkvw, CDIM, 0 * CDIM + h * HD, 32, 0,  kc, tid);  // Q rows
        stageB(sBhi, sBlo, qkvw, CDIM, 1 * CDIM + h * HD, 32, 32, kc, tid);  // K rows
        stageB(sBhi, sBlo, qkvw, CDIM, 2 * CDIM + h * HD, 32, 64, kc, tid);  // V rows
      }
      __syncthreads();
      gemm_step<6>(sAhi, sAlo, CDIM, m0, kc, sBhi, sBlo, qacc, lane);
    }

    // write q,k,v (+bias) for this head into sQKV  (cols: q 0..31, k 32..63, v 64..95)
#pragma unroll
    for (int t = 0; t < 6; ++t) {
      int col = t * 16 + (lane & 15);
      int p = col >> 5, d = col & 31;
      float bb = qkvb[p * CDIM + h * HD + d];
      int rbase = m0 + ((lane & 16) ? 8 : 0);
#pragma unroll
      for (int i = 0; i < 8; ++i) sQKV[(rbase + i) * 96 + col] = qacc[t][i] + bb;
    }
    __syncthreads();

    // A[d][e] = (sum_n feat(k[n][d]) * feat(v[n][e])) * gauss(d,e), 2 windows
    for (int i = tid; i < 2 * HD * HD; i += THREADS) {
      int w = i >> 10, rem = i & 1023, d = rem >> 5, e = rem & 31;
      float s = 0.f;
      for (int n = 0; n < NTOK; ++n) {
        float kk = sQKV[(w * 64 + n) * 96 + 32 + d];
        float vv = sQKV[(w * 64 + n) * 96 + 64 + e];
        s += featf(kk) * featf(vv);
      }
      float dd = (float)(d - e);
      sATT[i] = s * expf(-2.0f * dd * dd);                // sigma = 0.5
    }
    __syncthreads();

    // out[n][e] = (1-lam) * d^-0.5 * sum_d q[n][d] * A[d][e]
    float sc = sLAM[h] * QSCALE;
    for (int i = tid; i < 2 * NTOK * HD; i += THREADS) {
      int w = i / (NTOK * HD), rem = i % (NTOK * HD), n = rem >> 5, e = rem & 31;
      float s = 0.f;
      for (int d = 0; d < HD; ++d)
        s += sQKV[(w * 64 + n) * 96 + d] * sATT[w * 1024 + d * 32 + e];
      sF32[(w * 64 + n) * CDIM + h * HD + e] = s * sc;
    }
    __syncthreads();
  }

  // ---- stage 4: attn-out -> split-bf16 A operand
  for (int i = tid; i < ROWS * CDIM; i += THREADS) {
    int r = i / CDIM;
    float v = ((r & 63) < NTOK) ? sF32[i] : 0.0f;
    bf16 h, l; split2(v, h, l);
    sAhi[i] = h; sAlo[i] = l;
  }
  __syncthreads();

  // ---- stage 5: proj GEMM (K=192, N=192) + bias + shortcut -> x1 in sF32
  {
    v8f pacc[12];
#pragma unroll
    for (int t = 0; t < 12; ++t) pacc[t] = vzero8();
    for (int kc = 0; kc < CDIM; kc += 32) {
      __syncthreads();
#ifdef HAVE_TDM
      if (usepk) {
        tdm_stage(wpk + PK_PROJ + (size_t)(kc >> 5) * 12288, offBhi, offBlo, 6144, wave);
      } else
#endif
      {
        stageB(sBhi, sBlo, projw, CDIM, 0, 192, 0, kc, tid);
      }
      __syncthreads();
      gemm_step<12>(sAhi, sAlo, CDIM, m0, kc, sBhi, sBlo, pacc, lane);
    }
#pragma unroll
    for (int t = 0; t < 12; ++t) {
      int col = t * 16 + (lane & 15);
      int rbase = m0 + ((lane & 16) ? 8 : 0);
#pragma unroll
      for (int i = 0; i < 8; ++i) {
        int r = rbase + i;
        float v = 0.0f;
        if ((r & 63) < NTOK)
          v = pacc[t][i] + projb[col] + x[(size_t)row2tok(wpair, r) * CDIM + col];
        sF32[r * CDIM + col] = v;
      }
    }
  }
  __syncthreads();

  // ---- stage 6: LayerNorm2(x1) -> split-bf16 A operand (x1 stays in sF32)
  if (tid < ROWS) {
    float s = 0.f, ss = 0.f;
    for (int c = 0; c < CDIM; ++c) { float v = sF32[tid * CDIM + c]; s += v; ss += v * v; }
    float mu = s / CDIM;
    sMU[tid] = mu;
    sRS[tid] = 1.0f / sqrtf(ss / CDIM - mu * mu + 1e-5f);
  }
  __syncthreads();
  for (int i = tid; i < ROWS * CDIM; i += THREADS) {
    int r = i / CDIM, c = i % CDIM;
    float v = 0.0f;
    if ((r & 63) < NTOK) v = (sF32[i] - sMU[r]) * sRS[r] * n2w[c] + n2b[c];
    bf16 h, l; split2(v, h, l);
    sAhi[i] = h; sAlo[i] = l;
  }
  __syncthreads();

  // ---- stage 7: MLP  fc1(192->768) + exact GELU + fc2(768->192), K2 chunked by 64
  v8f macc[12];
#pragma unroll
  for (int t = 0; t < 12; ++t) macc[t] = vzero8();

  for (int kc2 = 0; kc2 < 4 * CDIM; kc2 += 64) {
    v8f hacc[4];
#pragma unroll
    for (int t = 0; t < 4; ++t) hacc[t] = vzero8();
    for (int kc = 0; kc < CDIM; kc += 32) {
      __syncthreads();
#ifdef HAVE_TDM
      if (usepk) {
        tdm_stage(wpk + PK_FC1 + (size_t)((kc2 >> 6) * 6 + (kc >> 5)) * 4096, offBhi, offBlo, 2048, wave);
      } else
#endif
      {
        stageB(sBhi, sBlo, f1w, CDIM, kc2, 64, 0, kc, tid);
      }
      __syncthreads();
      gemm_step<4>(sAhi, sAlo, CDIM, m0, kc, sBhi, sBlo, hacc, lane);
    }
    // bias + GELU(exact erf) -> split-bf16 hidden chunk (each wave owns its rows)
#pragma unroll
    for (int t = 0; t < 4; ++t) {
      int col = t * 16 + (lane & 15);
      float bb = f1b[kc2 + col];
      int rbase = m0 + ((lane & 16) ? 8 : 0);
#pragma unroll
      for (int i = 0; i < 8; ++i) {
        float v = hacc[t][i] + bb;
        v = 0.5f * v * (1.0f + erff(v * 0.70710678118654752f));
        bf16 hh, ll; split2(v, hh, ll);
        int idx = (rbase + i) * 64 + col;
        sHhi[idx] = hh; sHlo[idx] = ll;
      }
    }
    // fc2 partial: K chunk of 64 as two 32-K WMMA steps
    for (int kk = 0; kk < 64; kk += 32) {
      __syncthreads();
#ifdef HAVE_TDM
      if (usepk) {
        tdm_stage(wpk + PK_FC2 + (size_t)((kc2 >> 6) * 2 + (kk >> 5)) * 12288, offBhi, offBlo, 6144, wave);
      } else
#endif
      {
        stageB(sBhi, sBlo, f2w, 4 * CDIM, 0, 192, 0, kc2 + kk, tid);
      }
      __syncthreads();
      gemm_step<12>(sHhi, sHlo, 64, m0, kk, sBhi, sBlo, macc, lane);
    }
  }

  // ---- stage 8: out = x1 + fc2 + fc2_b, scatter back to (B, H*W, C)
#pragma unroll
  for (int t = 0; t < 12; ++t) {
    int col = t * 16 + (lane & 15);
    int rbase = m0 + ((lane & 16) ? 8 : 0);
#pragma unroll
    for (int i = 0; i < 8; ++i) {
      int r = rbase + i;
      if ((r & 63) < NTOK) {
        float v = sF32[r * CDIM + col] + macc[t][i] + f2b[col];
        out[(size_t)row2tok(wpair, r) * CDIM + col] = v;
      }
    }
  }
}

// ---------------- launcher ----------------

extern "C" void kernel_launch(void* const* d_in, const int* in_sizes, int n_in,
                              void* d_out, int out_size, void* d_ws, size_t ws_size,
                              hipStream_t stream) {
  (void)in_sizes; (void)n_in; (void)out_size;
  const float* x    = (const float*)d_in[0];
  // d_in[1] = attn_mask (unused by the module)
  const float* n1w  = (const float*)d_in[2];
  const float* n1b  = (const float*)d_in[3];
  const float* qkvw = (const float*)d_in[4];
  const float* qkvb = (const float*)d_in[5];
  const float* pw   = (const float*)d_in[6];
  const float* pb   = (const float*)d_in[7];
  const float* lk1  = (const float*)d_in[8];
  const float* lv1  = (const float*)d_in[9];
  const float* lk2  = (const float*)d_in[10];
  const float* lv2  = (const float*)d_in[11];
  const float* n2w  = (const float*)d_in[12];
  const float* n2b  = (const float*)d_in[13];
  const float* f1w  = (const float*)d_in[14];
  const float* f1b  = (const float*)d_in[15];
  const float* f2w  = (const float*)d_in[16];
  const float* f2b  = (const float*)d_in[17];
  float* out = (float*)d_out;

  int usepk = (d_ws != nullptr && ws_size >= PK_BYTES) ? 1 : 0;
  bf16* wpk = (bf16*)d_ws;

  if (usepk) {
    dim3 pgrid((PK_TOTAL / 2 + THREADS - 1) / THREADS);
    pack_weights<<<pgrid, dim3(THREADS), 0, stream>>>(qkvw, pw, f1w, f2w, wpk);
  }

  dim3 grid(2048);   // 4096 windows / 2 per workgroup
  dim3 block(THREADS);
  swin_diffattn_block<<<grid, block, 0, stream>>>(
      x, n1w, n1b, qkvw, qkvb, pw, pb, lk1, lv1, lk2, lv2,
      n2w, n2b, f1w, f1b, f2w, f2b, wpk, usepk, out);
}
